// DeformConvMmdet_61564061221316
// MI455X (gfx1250) — matrix operand order repaired and checked by reading, not compiled, not run on hardware
//
#include <hip/hip_runtime.h>
#include <hip/hip_bf16.h>
#include <math.h>
#include <stdint.h>

// ---------------------------------------------------------------------------
// Problem constants
// ---------------------------------------------------------------------------
#define BB   4
#define CC   256
#define HH   80
#define WW   80
#define HWC  (HH * WW)        // 6400
#define KT   9                // 3x3 taps
#define KDIM (CC * KT)        // 2304 reduction dim (k = tap*256 + c)

typedef _Float16 f16;
typedef __attribute__((ext_vector_type(16))) _Float16 v16h;
typedef __attribute__((ext_vector_type(8)))  float    v8f;
typedef __attribute__((ext_vector_type(4)))  float    v4f;
typedef __attribute__((ext_vector_type(4)))  int      v4i;

union AFrag { v16h v; v4f f[2]; };

// ---------------------------------------------------------------------------
// Kernel 0: repack deform_w [Co][C][3][3] (f32) -> wA [Co][K] (f16), K=tap*256+c
// ---------------------------------------------------------------------------
__global__ __launch_bounds__(256) void convert_w_kernel(
    const float* __restrict__ dw, f16* __restrict__ wA) {
  int idx = blockIdx.x * 256 + threadIdx.x;
  if (idx < CC * KDIM) {
    int co = idx / KDIM;
    int k  = idx % KDIM;
    int t  = k >> 8;          // tap
    int c  = k & 255;         // channel
    wA[idx] = (f16)dw[((size_t)co * CC + c) * KT + t];
  }
}

// ---------------------------------------------------------------------------
// Kernel 1: offset conv  (C=256 -> 27, 3x3, pad 1, bias), split into
// dy/dx/sigmoid(mask) planes, each [B][9][H*W] f32.
// Weight indices are wave-uniform -> scalar (SMEM) loads.
// ---------------------------------------------------------------------------
__global__ __launch_bounds__(128) void offset_conv_kernel(
    const float* __restrict__ x, const float* __restrict__ ow,
    const float* __restrict__ ob, float* __restrict__ dyA,
    float* __restrict__ dxA, float* __restrict__ mkA) {
  int n = blockIdx.x * 128 + threadIdx.x;   // pixel
  int b = blockIdx.y;
  int h = n / WW, w = n % WW;

  float acc[27];
#pragma unroll
  for (int o = 0; o < 27; ++o) acc[o] = ob[o];

  const float* xb = x + (size_t)b * CC * HWC;
  for (int c = 0; c < CC; ++c) {
    const float* xc = xb + (size_t)c * HWC;
    float xv[9];
#pragma unroll
    for (int j = 0; j < 9; ++j) {
      int yy = h + (j / 3) - 1;
      int xx = w + (j % 3) - 1;
      bool in = (yy >= 0) & (yy < HH) & (xx >= 0) & (xx < WW);
      xv[j] = in ? xc[yy * WW + xx] : 0.0f;
    }
    const float* wc = ow + (size_t)c * 9;   // ow[o*KDIM + c*9 + j]
#pragma unroll
    for (int o = 0; o < 27; ++o) {
#pragma unroll
      for (int j = 0; j < 9; ++j)
        acc[o] = fmaf(xv[j], wc[(size_t)o * KDIM + j], acc[o]);
    }
  }

#pragma unroll
  for (int o = 0; o < 27; ++o) {
    float z = acc[o];
    if (o < 9) {
      dyA[((size_t)b * 9 + o) * HWC + n] = z;
    } else if (o < 18) {
      dxA[((size_t)b * 9 + (o - 9)) * HWC + n] = z;
    } else {
      mkA[((size_t)b * 9 + (o - 18)) * HWC + n] = 1.0f / (1.0f + __expf(-z));
    }
  }
}

// ---------------------------------------------------------------------------
// Kernel 2: fused deformable-sampling + WMMA GEMM, double-buffered pipeline.
// Block tile: M=256 (all Co) x N=64 pixels. K loop: 72 steps of 32 (tap,cchunk).
// 8 waves: 4 (M, 64 rows each) x 2 (N, 32 cols each) -> 4x2 acc tiles/wave.
// A-tile staged via async global->LDS DMA (ASYNCcnt); B-tile gathered/bilinear.
// ---------------------------------------------------------------------------
#define BN    64
#define BK    32
#define APAD  40   // padded f16 row stride (80 B) to spread LDS banks
#define NSTEP 72   // 9 taps * 8 channel chunks

__global__ __launch_bounds__(256) void deform_gemm_kernel(
    const float* __restrict__ x, const f16* __restrict__ wA,
    const float* __restrict__ dyA, const float* __restrict__ dxA,
    const float* __restrict__ mkA, float* __restrict__ out) {
  __shared__ __align__(16) f16 As[2][256 * APAD];   // [m][k]
  __shared__ __align__(16) f16 Bs[2][BN * APAD];    // [n][k]  (K contiguous)

  const int tid  = threadIdx.x;
  const int n0   = blockIdx.x * BN;
  const int b    = blockIdx.y;
  const int lane = tid & 31;
  const int wv   = tid >> 5;       // wave 0..7
  const int wm   = wv & 3;         // M group (64 rows)
  const int wn   = wv >> 2;        // N group (32 cols)
  const int lh   = lane & 15;
  const int hi   = lane >> 4;      // 0/1: lane half selects K sub-range

  const int p  = tid & 63;         // pixel within tile (4x redundant owners)
  const int cg = tid >> 6;         // 0..3 -> 8 channels each

  v8f acc[4][2] = {};

  const float* xb = x + (size_t)b * CC * HWC;

  // per-thread bilinear params for (pixel p, current prefetch tap)
  v4i id4;
  v4f wg4;
  float mv = 0.0f;
  float gv[8];                     // gathered (pre-mask) values for next step

  auto compute_params = [&](int t) {
    int n = n0 + p;
    int h = n / WW, w = n % WW;
    size_t oo = ((size_t)b * KT + t) * HWC + n;
    float dy = dyA[oo], dx = dxA[oo];
    mv = mkA[oo];
    float py = (float)(h + (t / 3) - 1) + dy;
    float px = (float)(w + (t % 3) - 1) + dx;
    float fy0 = floorf(py), fx0 = floorf(px);
    int iy0 = (int)fy0, ix0 = (int)fx0;
    int iy1 = iy0 + 1,  ix1 = ix0 + 1;
    float wy1 = py - fy0, wx1 = px - fx0;
    float wy0 = 1.0f - wy1, wx0 = 1.0f - wx1;
    float vy0 = (iy0 >= 0 && iy0 < HH) ? 1.0f : 0.0f;
    float vy1 = (iy1 >= 0 && iy1 < HH) ? 1.0f : 0.0f;
    float vx0 = (ix0 >= 0 && ix0 < WW) ? 1.0f : 0.0f;
    float vx1 = (ix1 >= 0 && ix1 < WW) ? 1.0f : 0.0f;
    int cy0 = min(max(iy0, 0), HH - 1), cy1 = min(max(iy1, 0), HH - 1);
    int cx0 = min(max(ix0, 0), WW - 1), cx1 = min(max(ix1, 0), WW - 1);
    id4.x = cy0 * WW + cx0; id4.y = cy0 * WW + cx1;
    id4.z = cy1 * WW + cx0; id4.w = cy1 * WW + cx1;
    wg4.x = wy0 * wx0 * vy0 * vx0; wg4.y = wy0 * wx1 * vy0 * vx1;
    wg4.z = wy1 * wx0 * vy1 * vx0; wg4.w = wy1 * wx1 * vy1 * vx1;
  };

  // async DMA of one A-tile row (32 f16 = 64 B) straight into LDS[bsel]
  auto copy_a_async = [&](int kg, int bsel) {
    const f16* src = wA + (size_t)tid * KDIM + kg;
    f16* dst = &As[bsel][tid * APAD];
    uint64_t ga = (uint64_t)(uintptr_t)src;          // global VA
    uint32_t la = (uint32_t)(uintptr_t)dst;          // LDS byte offset (addr[31:0])
    asm volatile(
        "global_load_async_to_lds_b128 %0, %1, off\n\t"
        "global_load_async_to_lds_b128 %0, %1, off offset:16\n\t"
        "global_load_async_to_lds_b128 %0, %1, off offset:32\n\t"
        "global_load_async_to_lds_b128 %0, %1, off offset:48"
        :: "v"(la), "v"(ga) : "memory");
  };

  auto gather = [&](int c0) {
    const float* xp = xb + (size_t)(c0 + cg * 8) * HWC;
#pragma unroll
    for (int j = 0; j < 8; ++j) {
      const float* q = xp + (size_t)j * HWC;
      gv[j] = q[id4.x] * wg4.x + q[id4.y] * wg4.y +
              q[id4.z] * wg4.z + q[id4.w] * wg4.w;
    }
  };

  auto store_b = [&](int bsel) {
    union { f16 h[8]; v4f f; } buf;
#pragma unroll
    for (int j = 0; j < 8; ++j) buf.h[j] = (f16)(gv[j] * mv);
    *(v4f*)&Bs[bsel][p * APAD + cg * 8] = buf.f;
  };

  // ---- prologue: fill buffer 0 with step 0 (tap 0, chunk 0) ----
  compute_params(0);
  copy_a_async(0, 0);
  gather(0);
  store_b(0);
  asm volatile("s_wait_asynccnt 0x0" ::: "memory");
  __syncthreads();

  // ---- pipelined main loop: one barrier per K-step ----
  for (int step = 0; step < NSTEP; ++step) {
    const int cur  = step & 1;
    const int nxt  = cur ^ 1;
    const int next = step + 1;

    // 1) issue next step's long-latency work first
    if (next < NSTEP) {
      const int tn  = next >> 3;
      const int ccn = next & 7;
      if (ccn == 0) compute_params(tn);
      copy_a_async(tn * CC + ccn * BK, nxt);   // async DMA -> LDS[nxt]
      gather(ccn * BK);                        // global gathers -> regs
    }

    // 2) fragments + 4x2 WMMA on current buffer
    const f16* AsC = As[cur];
    const f16* BsC = Bs[cur];
    AFrag af[4], bf[2];
#pragma unroll
    for (int i = 0; i < 4; ++i) {
      int row = wm * 64 + i * 16 + lh;
      const f16* q = &AsC[row * APAD + hi * 8];
      af[i].f[0] = *(const v4f*)(q);
      af[i].f[1] = *(const v4f*)(q + 16);
    }
#pragma unroll
    for (int j = 0; j < 2; ++j) {
      int nn = wn * 32 + j * 16 + lh;
      const f16* q = &BsC[nn * APAD + hi * 8];
      bf[j].f[0] = *(const v4f*)(q);
      bf[j].f[1] = *(const v4f*)(q + 16);
    }
#pragma unroll
    for (int i = 0; i < 4; ++i)
#pragma unroll
      for (int j = 0; j < 2; ++j)
        acc[i][j] = __builtin_amdgcn_wmma_f32_16x16x32_f16(
            false, af[i].v, false, bf[j].v, (short)0, acc[i][j],
            false, false);

    // 3) finish next step's B tile (modulate + cvt + LDS store)
    if (next < NSTEP) store_b(nxt);

    // 4) all async DMA for LDS[nxt] must land before anyone reads it
    asm volatile("s_wait_asynccnt 0x0" ::: "memory");
    __syncthreads();
  }

  // --- epilogue: C/D layout -> out[b][co][h*w] f32 ---
  float* ob = out + (size_t)b * CC * HWC;
#pragma unroll
  for (int i = 0; i < 4; ++i) {
#pragma unroll
    for (int j = 0; j < 2; ++j) {
      int col = n0 + wn * 32 + j * 16 + lh;
#pragma unroll
      for (int r = 0; r < 8; ++r) {
        int row = wm * 64 + i * 16 + r + (hi << 3);
        ob[(size_t)row * HWC + col] = acc[i][j][r];
      }
    }
  }
}

// ---------------------------------------------------------------------------
// Launch
// ---------------------------------------------------------------------------
extern "C" void kernel_launch(void* const* d_in, const int* in_sizes, int n_in,
                              void* d_out, int out_size, void* d_ws, size_t ws_size,
                              hipStream_t stream) {
  const float* x  = (const float*)d_in[0];
  const float* ow = (const float*)d_in[1];   // [27][256][3][3]
  const float* ob = (const float*)d_in[2];   // [27]
  const float* dw = (const float*)d_in[3];   // [256][256][3][3]
  float* out = (float*)d_out;

  char* ws = (char*)d_ws;
  const size_t planeN = (size_t)BB * KT * HWC;   // 230400 floats
  float* dyA = (float*)ws;
  float* dxA = dyA + planeN;
  float* mkA = dxA + planeN;
  f16* wA = (f16*)(ws + 3 * planeN * sizeof(float));   // 256*2304 f16

  // 0) repack weights to f16 [Co][K]
  convert_w_kernel<<<(CC * KDIM + 255) / 256, 256, 0, stream>>>(dw, wA);
  // 1) offset conv -> dy / dx / sigmoid(mask)
  offset_conv_kernel<<<dim3(HWC / 128, BB), 128, 0, stream>>>(
      x, ow, ob, dyA, dxA, mkA);
  // 2) fused deformable sampling + WMMA GEMM (async LDS pipeline)
  deform_gemm_kernel<<<dim3(HWC / BN, BB), 256, 0, stream>>>(
      x, wA, dyA, dxA, mkA, out);
}
